// LayerPool_76373108457726
// MI455X (gfx1250) — matrix operand order repaired and checked by reading, not compiled, not run on hardware
//
#include <hip/hip_runtime.h>
#include <hip/hip_bf16.h>
#include <math.h>

// ---------------------------------------------------------------------------
// Routed transformer block (pre-LN attention + pre-LN FFN), f16 WMMA compute,
// f32 accumulation, flash-attention with online softmax.
// Target: gfx1250 (CDNA5, wave32, V_WMMA_F32_16X16X32_F16,
//          GLOBAL_LOAD_ASYNC_TO_LDS_B128 double-buffered GEMM staging).
// ---------------------------------------------------------------------------

typedef __attribute__((ext_vector_type(16))) _Float16 v16h;
typedef __attribute__((ext_vector_type(8)))  _Float16 v8h;
typedef __attribute__((ext_vector_type(8)))  float    v8f;
typedef __attribute__((ext_vector_type(4)))  int      v4i;

union F16x16 { v16h v; v8h h[2]; };

static constexpr int NL  = 8;
static constexpr int NB  = 8;
static constexpr int NS  = 2048;
static constexpr int ND  = 1024;
static constexpr int NH  = 16;
static constexpr int NDH = 64;
static constexpr int NF  = 4096;

static __device__ __forceinline__ v8f wmma_f16(const v16h& a, const v16h& b, v8f c) {
    return __builtin_amdgcn_wmma_f32_16x16x32_f16(false, a, false, b, (short)0, c,
                                                  false, false);
}

// Async global -> LDS copy (CDNA5). Per compile probe, the builtin's param 0 is
// a pointer-to-global int4 (v4i32); mirror that for the LDS destination.
#if __has_builtin(__builtin_amdgcn_global_load_async_to_lds_b128) && \
    __has_builtin(__builtin_amdgcn_s_wait_asynccnt)
#define ASYNC_LDS 1
typedef __attribute__((address_space(1))) v4i* gv4i_p;
typedef __attribute__((address_space(3))) v4i* lv4i_p;
static __device__ __forceinline__ void async_cp16(void* lds, const void* g) {
    __builtin_amdgcn_global_load_async_to_lds_b128((gv4i_p)g, (lv4i_p)lds, 0, 0);
}
#else
#define ASYNC_LDS 0
#endif

// ---------------------------------------------------------------------------
// Gather routed weight [idx[b]], convert f32 -> f16, transpose: out[b][n][k].
// LDS-tiled 32x32 so both the global read and the global write are coalesced.
// ---------------------------------------------------------------------------
__global__ __launch_bounds__(256)
void wconv_kernel(const float* __restrict__ W, const int* __restrict__ idx,
                  _Float16* __restrict__ out, int K, int N) {
    __shared__ float t[32][33];
    const int b  = blockIdx.z;
    const int k0 = blockIdx.x * 32, n0 = blockIdx.y * 32;
    const int tx = threadIdx.x & 31, ty = threadIdx.x >> 5;   // 32 x 8
    const float* src = W + (size_t)idx[b] * K * N;
    for (int i = ty; i < 32; i += 8)
        t[i][tx] = src[(size_t)(k0 + i) * N + n0 + tx];
    __syncthreads();
    _Float16* dst = out + (size_t)b * N * K;
    for (int i = ty; i < 32; i += 8)
        dst[(size_t)(n0 + i) * K + k0 + tx] = (_Float16)t[tx][i];
}

// ---------------------------------------------------------------------------
// LayerNorm over D=1024, routed scale/bias, emits f16. One block per row.
// ---------------------------------------------------------------------------
__global__ void ln_kernel(const float* __restrict__ x, const int* __restrict__ idx,
                          const float* __restrict__ sc, const float* __restrict__ bi,
                          _Float16* __restrict__ out) {
    __shared__ float red[256];
    int row   = blockIdx.x;            // over NB*NS
    int b     = row / NS;
    int layer = idx[b];
    const float* xr = x + (size_t)row * ND;

    float lsum = 0.f, lsq = 0.f;
    for (int d = threadIdx.x; d < ND; d += 256) {
        float v = xr[d]; lsum += v; lsq += v * v;
    }
    red[threadIdx.x] = lsum; __syncthreads();
    for (int o = 128; o > 0; o >>= 1) {
        if (threadIdx.x < o) red[threadIdx.x] += red[threadIdx.x + o];
        __syncthreads();
    }
    float mean = red[0] * (1.0f / ND);
    __syncthreads();
    red[threadIdx.x] = lsq; __syncthreads();
    for (int o = 128; o > 0; o >>= 1) {
        if (threadIdx.x < o) red[threadIdx.x] += red[threadIdx.x + o];
        __syncthreads();
    }
    float var = red[0] * (1.0f / ND) - mean * mean;
    float inv = rsqrtf(var + 1e-5f);

    const float* sg = sc + (size_t)layer * ND;
    const float* bg = bi + (size_t)layer * ND;
    _Float16*    og = out + (size_t)row * ND;
    for (int d = threadIdx.x; d < ND; d += 256)
        og[d] = (_Float16)((xr[d] - mean) * inv * sg[d] + bg[d]);
}

// ---------------------------------------------------------------------------
// WMMA GEMM: C[b] = A[b] (MxK, f16 row-major) * Bt[b] (NxK, f16 "transposed")
// Epilogues: 0 = +bias -> f16 ; 1 = gelu(+bias) -> f16 ; 2 = resid + +bias -> f32
// Block: 128x128 tile, 256 threads (8 waves), wave = 32x64 (2x4 WMMA tiles),
// double-buffered LDS with async global->LDS prefetch of the next K-slice.
// ---------------------------------------------------------------------------
template <int EPI>
__global__ __launch_bounds__(256)
void gemm_kernel(const _Float16* __restrict__ A, const _Float16* __restrict__ Bt,
                 const float* __restrict__ bias, const int* __restrict__ idx,
                 const float* __restrict__ resid, void* __restrict__ out,
                 int M, int N, int K) {
    constexpr int BM = 128, BN = 128, BK = 32;
    constexpr int AST = BK + 8;                        // 40 halves = 80B (16B-mult)
    __shared__ __align__(16) _Float16 As[2][BM * AST];
    __shared__ __align__(16) _Float16 Bs[2][BN * AST];

    const int b    = blockIdx.z;
    const int bm   = blockIdx.x, bn = blockIdx.y;
    const int tid  = threadIdx.x;
    const int lane = tid & 31, wave = tid >> 5;
    const int wm   = wave >> 1, wn = wave & 1;         // 4 x 2 wave grid
    const int lm   = lane & 15, lh = lane >> 4;

    const _Float16* Ab = A  + (size_t)b * M * K + (size_t)(bm * BM) * K;
    const _Float16* Bb = Bt + (size_t)b * N * K + (size_t)(bn * BN) * K;

    v8f c[2][4] = {};

    // Each K-slice is BM x BK halves = 512 chunks of 8 halves; 2 chunks/thread.
    auto compute = [&](int buf) {
        F16x16 a[2], bf[4];
        for (int tm = 0; tm < 2; ++tm) {
            int m = wm * 32 + tm * 16 + lm;
            a[tm].h[0] = *(const v8h*)&As[buf][m * AST + lh * 8];
            a[tm].h[1] = *(const v8h*)&As[buf][m * AST + lh * 8 + 16];
        }
        for (int tn = 0; tn < 4; ++tn) {
            int n = wn * 64 + tn * 16 + lm;
            bf[tn].h[0] = *(const v8h*)&Bs[buf][n * AST + lh * 8];
            bf[tn].h[1] = *(const v8h*)&Bs[buf][n * AST + lh * 8 + 16];
        }
        for (int tm = 0; tm < 2; ++tm)
            for (int tn = 0; tn < 4; ++tn)
                c[tm][tn] = wmma_f16(a[tm].v, bf[tn].v, c[tm][tn]);
    };

#if ASYNC_LDS
    auto stage = [&](int buf, int k0) {
        for (int i = 0; i < 2; ++i) {
            int c8 = tid + i * 256;                    // 512 chunks
            int r = c8 >> 2, cc = c8 & 3;
            async_cp16(&As[buf][r * AST + cc * 8], &Ab[(size_t)r * K + k0 + cc * 8]);
            async_cp16(&Bs[buf][r * AST + cc * 8], &Bb[(size_t)r * K + k0 + cc * 8]);
        }
    };
    stage(0, 0);
    __builtin_amdgcn_s_wait_asynccnt(0);
    __syncthreads();
    int buf = 0;
    for (int k0 = 0; k0 < K; k0 += BK, buf ^= 1) {
        if (k0 + BK < K) stage(buf ^ 1, k0 + BK);      // prefetch next slice
        compute(buf);
        __builtin_amdgcn_s_wait_asynccnt(0);
        __syncthreads();
    }
#else
    v8h ra[2], rb[2];
    auto load_regs = [&](int k0) {
        for (int i = 0; i < 2; ++i) {
            int c8 = tid + i * 256;
            int r = c8 >> 2, cc = c8 & 3;
            ra[i] = *(const v8h*)&Ab[(size_t)r * K + k0 + cc * 8];
            rb[i] = *(const v8h*)&Bb[(size_t)r * K + k0 + cc * 8];
        }
    };
    auto store_regs = [&](int buf) {
        for (int i = 0; i < 2; ++i) {
            int c8 = tid + i * 256;
            int r = c8 >> 2, cc = c8 & 3;
            *(v8h*)&As[buf][r * AST + cc * 8] = ra[i];
            *(v8h*)&Bs[buf][r * AST + cc * 8] = rb[i];
        }
    };
    load_regs(0);
    store_regs(0);
    __syncthreads();
    int buf = 0;
    for (int k0 = 0; k0 < K; k0 += BK, buf ^= 1) {
        bool pf = (k0 + BK) < K;
        if (pf) load_regs(k0 + BK);
        compute(buf);
        if (pf) store_regs(buf ^ 1);
        __syncthreads();
    }
#endif

    const int layer = idx[b];
    for (int tm = 0; tm < 2; ++tm)
        for (int tn = 0; tn < 4; ++tn) {
            int   n  = bn * BN + wn * 64 + tn * 16 + lm;
            float bv = bias[(size_t)layer * N + n];
            for (int r = 0; r < 8; ++r) {
                int    m = bm * BM + wm * 32 + tm * 16 + r + lh * 8;
                float  v = c[tm][tn][r] + bv;
                size_t o = (size_t)b * M * N + (size_t)m * N + n;
                if (EPI == 0) {
                    ((_Float16*)out)[o] = (_Float16)v;
                } else if (EPI == 1) {
                    float g = 0.5f * v *
                              (1.f + tanhf(0.7978845608f * (v + 0.044715f * v * v * v)));
                    ((_Float16*)out)[o] = (_Float16)g;
                } else {
                    ((float*)out)[o] = resid[o] + v;
                }
            }
        }
}

// ---------------------------------------------------------------------------
// Flash attention: per (qblock of 64 rows, head, batch). f16 WMMA, f32 softmax.
// q/k/v/out: [B,S,D] f16, head h occupies columns [h*64, h*64+64).
// ---------------------------------------------------------------------------
__global__ __launch_bounds__(128)
void attn_kernel(const _Float16* __restrict__ q, const _Float16* __restrict__ k,
                 const _Float16* __restrict__ v, _Float16* __restrict__ o) {
    constexpr int QST = 72;                           // half stride (144B)
    __shared__ __align__(16) _Float16 Qs[64 * QST];
    __shared__ __align__(16) _Float16 Ks[64 * QST];   // Bt layout: [key][dh]
    __shared__ __align__(16) _Float16 Vs[64 * QST];   // Bt layout: [dh][key]
    __shared__ __align__(16) _Float16 Ps[64 * QST];
    __shared__ float Sc[64 * 68];
    __shared__ float mrun[64], lrun[64], alpha[64], red2[64][2];

    const int qb = blockIdx.x, h = blockIdx.y, b = blockIdx.z;
    const int tid = threadIdx.x;
    const int lane = tid & 31, wave = tid >> 5;
    const int wm = wave >> 1, wn = wave & 1;
    const int lm = lane & 15, lh = lane >> 4;

    const _Float16* qbase = q + (size_t)b * NS * ND + (size_t)qb * 64 * ND + h * NDH;
    const _Float16* kbase = k + (size_t)b * NS * ND + h * NDH;
    const _Float16* vbase = v + (size_t)b * NS * ND + h * NDH;

    // stage Q once, scaled by 1/sqrt(DH) = 0.125
    for (int c8 = tid; c8 < 512; c8 += 128) {
        int r = c8 >> 3, cc = c8 & 7;
        v8h t = *(const v8h*)&qbase[(size_t)r * ND + cc * 8];
        _Float16* dst = &Qs[r * QST + cc * 8];
        for (int j = 0; j < 8; ++j) dst[j] = (_Float16)((float)t[j] * 0.125f);
    }
    if (tid < 64) { mrun[tid] = -1e30f; lrun[tid] = 0.f; }
    v8f oacc[2][2] = {};
    __syncthreads();

    for (int kb = 0; kb < NS / 64; ++kb) {
        const _Float16* kt = kbase + (size_t)kb * 64 * ND;
        const _Float16* vt = vbase + (size_t)kb * 64 * ND;
        for (int c8 = tid; c8 < 512; c8 += 128) {
            int r = c8 >> 3, cc = c8 & 7;
            *(v8h*)&Ks[r * QST + cc * 8] = *(const v8h*)&kt[(size_t)r * ND + cc * 8];
            v8h tv = *(const v8h*)&vt[(size_t)r * ND + cc * 8];
            for (int j = 0; j < 8; ++j) Vs[(cc * 8 + j) * QST + r] = tv[j];
        }
        __syncthreads();

        // S = Q * K^T   (M=64 queries, N=64 keys, K=DH=64)
        v8f s[2][2] = {};
        for (int ks = 0; ks < 64; ks += 32) {
            F16x16 a[2], bf[2];
            for (int tm = 0; tm < 2; ++tm) {
                int m = wm * 32 + tm * 16 + lm;
                a[tm].h[0] = *(const v8h*)&Qs[m * QST + ks + lh * 8];
                a[tm].h[1] = *(const v8h*)&Qs[m * QST + ks + lh * 8 + 16];
            }
            for (int tn = 0; tn < 2; ++tn) {
                int n = wn * 32 + tn * 16 + lm;
                bf[tn].h[0] = *(const v8h*)&Ks[n * QST + ks + lh * 8];
                bf[tn].h[1] = *(const v8h*)&Ks[n * QST + ks + lh * 8 + 16];
            }
            for (int tm = 0; tm < 2; ++tm)
                for (int tn = 0; tn < 2; ++tn)
                    s[tm][tn] = wmma_f16(a[tm].v, bf[tn].v, s[tm][tn]);
        }
        for (int tm = 0; tm < 2; ++tm)
            for (int tn = 0; tn < 2; ++tn) {
                int n = wn * 32 + tn * 16 + lm;
                for (int r = 0; r < 8; ++r) {
                    int m = wm * 32 + tm * 16 + r + lh * 8;
                    Sc[m * 68 + n] = s[tm][tn][r];
                }
            }
        __syncthreads();

        // online softmax: 2 threads per row, 32 cols each
        int row = tid >> 1, half = tid & 1;
        float lmax = -1e30f;
        for (int j = 0; j < 32; ++j)
            lmax = fmaxf(lmax, Sc[row * 68 + half * 32 + j]);
        red2[row][half] = lmax;
        __syncthreads();
        if (half == 0) {
            float nm = fmaxf(mrun[row], fmaxf(red2[row][0], red2[row][1]));
            alpha[row] = __expf(mrun[row] - nm);
            mrun[row]  = nm;
        }
        __syncthreads();
        float nm = mrun[row], ls = 0.f;
        for (int j = 0; j < 32; ++j) {
            float p = __expf(Sc[row * 68 + half * 32 + j] - nm);
            ls += p;
            Ps[row * QST + half * 32 + j] = (_Float16)p;
        }
        red2[row][half] = ls;
        __syncthreads();
        if (half == 0) lrun[row] = lrun[row] * alpha[row] + red2[row][0] + red2[row][1];
        __syncthreads();

        // rescale O accumulators, then O += P * V  (A=P [q x key], Bt=Vs [dh][key])
        for (int tm = 0; tm < 2; ++tm)
            for (int r = 0; r < 8; ++r) {
                int   m  = wm * 32 + tm * 16 + r + lh * 8;
                float al = alpha[m];
                oacc[tm][0][r] *= al;
                oacc[tm][1][r] *= al;
            }
        for (int ks = 0; ks < 64; ks += 32) {
            F16x16 a[2], bf[2];
            for (int tm = 0; tm < 2; ++tm) {
                int m = wm * 32 + tm * 16 + lm;
                a[tm].h[0] = *(const v8h*)&Ps[m * QST + ks + lh * 8];
                a[tm].h[1] = *(const v8h*)&Ps[m * QST + ks + lh * 8 + 16];
            }
            for (int tn = 0; tn < 2; ++tn) {
                int n = wn * 32 + tn * 16 + lm;
                bf[tn].h[0] = *(const v8h*)&Vs[n * QST + ks + lh * 8];
                bf[tn].h[1] = *(const v8h*)&Vs[n * QST + ks + lh * 8 + 16];
            }
            for (int tm = 0; tm < 2; ++tm)
                for (int tn = 0; tn < 2; ++tn)
                    oacc[tm][tn] = wmma_f16(a[tm].v, bf[tn].v, oacc[tm][tn]);
        }
        __syncthreads();
    }

    _Float16* obase = o + (size_t)b * NS * ND + (size_t)qb * 64 * ND + h * NDH;
    for (int tm = 0; tm < 2; ++tm)
        for (int tn = 0; tn < 2; ++tn) {
            int n = wn * 32 + tn * 16 + lm;
            for (int r = 0; r < 8; ++r) {
                int   m   = wm * 32 + tm * 16 + r + lh * 8;
                float den = lrun[m];
                obase[(size_t)m * ND + n] = (_Float16)(oacc[tm][tn][r] / den);
            }
        }
}

// ---------------------------------------------------------------------------
extern "C" void kernel_launch(void* const* d_in, const int* in_sizes, int n_in,
                              void* d_out, int out_size, void* d_ws, size_t ws_size,
                              hipStream_t stream) {
    (void)in_sizes; (void)n_in; (void)out_size; (void)ws_size;
    const float* x     = (const float*)d_in[0];
    const int*   idx   = (const int*)d_in[1];
    const float* Wq    = (const float*)d_in[2];
    const float* bq    = (const float*)d_in[3];
    const float* Wk    = (const float*)d_in[4];
    const float* bk    = (const float*)d_in[5];
    const float* Wv    = (const float*)d_in[6];
    const float* bv    = (const float*)d_in[7];
    const float* Wo    = (const float*)d_in[8];
    const float* bo    = (const float*)d_in[9];
    const float* ln1_s = (const float*)d_in[10];
    const float* ln1_b = (const float*)d_in[11];
    const float* ln2_s = (const float*)d_in[12];
    const float* ln2_b = (const float*)d_in[13];
    const float* W1    = (const float*)d_in[14];
    const float* b1    = (const float*)d_in[15];
    const float* W2    = (const float*)d_in[16];
    const float* b2    = (const float*)d_in[17];
    float*       out   = (float*)d_out;

    _Float16* ws = (_Float16*)d_ws;
    const size_t SD = (size_t)NB * NS * ND;   // 16M halves
    const size_t SF = (size_t)NB * NS * NF;   // 64M halves
    const size_t WD = (size_t)NB * ND * ND;   // 8M halves
    const size_t WF = (size_t)NB * ND * NF;   // 32M halves
    _Float16* h16 = ws;
    _Float16* q16 = ws + SD;
    _Float16* k16 = ws + 2 * SD;
    _Float16* v16 = ws + 3 * SD;
    _Float16* a16 = ws + 4 * SD;
    _Float16* h1  = ws + 5 * SD;
    _Float16* wqT = h1 + SF;
    _Float16* wkT = wqT + WD;
    _Float16* wvT = wkT + WD;
    _Float16* woT = wvT + WD;
    _Float16* w1T = woT + WD;
    _Float16* w2T = w1T + WF;

    // --- gather + convert + transpose routed weights to f16 [b][n][k] ---
    dim3 ct(256);
    wconv_kernel<<<dim3(ND / 32, ND / 32, NB), ct, 0, stream>>>(Wq, idx, wqT, ND, ND);
    wconv_kernel<<<dim3(ND / 32, ND / 32, NB), ct, 0, stream>>>(Wk, idx, wkT, ND, ND);
    wconv_kernel<<<dim3(ND / 32, ND / 32, NB), ct, 0, stream>>>(Wv, idx, wvT, ND, ND);
    wconv_kernel<<<dim3(ND / 32, ND / 32, NB), ct, 0, stream>>>(Wo, idx, woT, ND, ND);
    wconv_kernel<<<dim3(ND / 32, NF / 32, NB), ct, 0, stream>>>(W1, idx, w1T, ND, NF);
    wconv_kernel<<<dim3(NF / 32, ND / 32, NB), ct, 0, stream>>>(W2, idx, w2T, NF, ND);

    dim3 blk(256);
    dim3 gD(NS / 128, ND / 128, NB);
    dim3 gF(NS / 128, NF / 128, NB);

    // --- attention (pre-LN) ---
    ln_kernel<<<NB * NS, 256, 0, stream>>>(x, idx, ln1_s, ln1_b, h16);
    gemm_kernel<0><<<gD, blk, 0, stream>>>(h16, wqT, bq, idx, nullptr, q16, NS, ND, ND);
    gemm_kernel<0><<<gD, blk, 0, stream>>>(h16, wkT, bk, idx, nullptr, k16, NS, ND, ND);
    gemm_kernel<0><<<gD, blk, 0, stream>>>(h16, wvT, bv, idx, nullptr, v16, NS, ND, ND);
    attn_kernel<<<dim3(NS / 64, NH, NB), dim3(128), 0, stream>>>(q16, k16, v16, a16);
    gemm_kernel<2><<<gD, blk, 0, stream>>>(a16, woT, bo, idx, x, out, NS, ND, ND);

    // --- FFN (pre-LN) ---
    ln_kernel<<<NB * NS, 256, 0, stream>>>(out, idx, ln2_s, ln2_b, h16);
    gemm_kernel<1><<<gF, blk, 0, stream>>>(h16, w1T, b1, idx, nullptr, h1, NS, NF, ND);
    gemm_kernel<2><<<gD, blk, 0, stream>>>(h1, w2T, b2, idx, out, out, NS, ND, NF);
}
